// SelfAttentionBlock_Vit_65283502899743
// MI455X (gfx1250) — compile-verified, matrix operand order checked by reading, not run on hardware
//
#include <hip/hip_runtime.h>

// MI455X / gfx1250: wave32, WMMA 16x16x32 bf16 with f32 accum, async
// global->LDS tile staging (ASYNCcnt). Flash-attention ViT block.
// Workspace layout (needs ~21 MB of d_ws):
//   [ 0MB) xnorm bf16 (8192x256)   [ 4MB) q bf16 (B,nh,N,hd)
//   [ 8MB) k bf16 (B,nh,N,hd)      [12MB) v^T bf16 (B,nh,hd,N)
//   [16MB) attn_out bf16 (8192x256)[20MB) qkv_w bf16 | proj_w bf16

typedef unsigned short u16;
typedef __bf16 bf16_t;
typedef bf16_t v8bf  __attribute__((ext_vector_type(8)));
typedef bf16_t v16bf __attribute__((ext_vector_type(16)));
typedef float  v8f   __attribute__((ext_vector_type(8)));

#define NTOK   4096
#define CCH    256
#define NH     8
#define HD     32
#define BATCH  2
#define LOG2E  1.44269504088896340736f
#define QK_SCALE 0.17677669529663688f   // 32^-0.5

__device__ __forceinline__ u16 f2bf(float f) {
    union { float f; unsigned u; } v; v.f = f;
    unsigned r = v.u + 0x7FFFu + ((v.u >> 16) & 1u);
    return (u16)(r >> 16);
}

// Build a 16x32 bf16 A/B fragment for v_wmma_f32_16x16x32_bf16.
// Per ISA 7.12.2: lane holds K = {kb..kb+7} and {16+kb..16+kb+7}, kb = (lane>>4)*8.
__device__ __forceinline__ v16bf ldfrag(const u16* p, int kb) {
    v8bf lo = *(const v8bf*)(p + kb);
    v8bf hi = *(const v8bf*)(p + 16 + kb);
    return __builtin_shufflevector(lo, hi, 0,1,2,3,4,5,6,7,8,9,10,11,12,13,14,15);
}

// CDNA5 async DMA: 16B global -> LDS, no VGPR round-trip, tracked by ASYNCcnt.
// LDS operand is the low 32 bits of the generic pointer.
__device__ __forceinline__ void async_cp16(u16* lds_dst, const u16* gsrc) {
    unsigned loff = (unsigned)(size_t)lds_dst;
    asm volatile("global_load_async_to_lds_b128 %0, %1, off"
                 :: "v"(loff), "v"(gsrc) : "memory");
}
__device__ __forceinline__ void wait_async0() {
    asm volatile("s_wait_asynccnt 0x0" ::: "memory");
}

// ---------------- kernel 0: fp32 weights -> bf16 ----------------
__global__ void __launch_bounds__(256) vit_cvt_w(const float* __restrict__ qw,
                                                 const float* __restrict__ pw,
                                                 u16* __restrict__ qwb,
                                                 u16* __restrict__ pwb) {
    int i = blockIdx.x * 256 + threadIdx.x;      // grid covers 768*256
    if (i < 768 * 256) qwb[i] = f2bf(qw[i]);
    if (i < 256 * 256) pwb[i] = f2bf(pw[i]);
}

// ---------------- kernel 1: transpose + LayerNorm -> bf16 -------
// x is (B, C, N); LN over C per token n. 32 tokens per block, coalesced
// loads along N, LDS transpose (pad 257 to dodge bank conflicts), one wave
// per token for the reduction.
__global__ void __launch_bounds__(256) vit_ln(const float* __restrict__ x,
                                              const float* __restrict__ nw,
                                              const float* __restrict__ nb,
                                              u16* __restrict__ xn) {
    __shared__ float tile[32 * 257];
    int t = threadIdx.x;
    int bid = blockIdx.x;
    int b  = bid >> 7;                  // 128 blocks per batch
    int n0 = (bid & 127) * 32;
    const float* xb = x + (size_t)b * CCH * NTOK;
    for (int c0 = 0; c0 < CCH; c0 += 8) {
        int c = c0 + (t >> 5);
        int n = t & 31;
        tile[n * 257 + c] = xb[(size_t)c * NTOK + n0 + n];
    }
    __syncthreads();
    int w = t >> 5, l = t & 31;
    for (int jj = 0; jj < 4; ++jj) {
        int j = w * 4 + jj;
        const float* row = &tile[j * 257];
        float s = 0.f, ss = 0.f;
        #pragma unroll
        for (int kk = 0; kk < 8; ++kk) {
            float v = row[kk * 32 + l];
            s += v; ss += v * v;
        }
        #pragma unroll
        for (int m = 1; m < 32; m <<= 1) {
            s  += __shfl_xor(s,  m);
            ss += __shfl_xor(ss, m);
        }
        float mu   = s * (1.f / 256.f);
        float var  = ss * (1.f / 256.f) - mu * mu;
        float rstd = rsqrtf(var + 1e-5f);
        u16* dst = xn + ((size_t)b * NTOK + n0 + j) * CCH;
        #pragma unroll
        for (int kk = 0; kk < 8; ++kk) {
            int c = kk * 32 + l;
            dst[c] = f2bf((row[c] - mu) * rstd * nw[c] + nb[c]);
        }
    }
}

// ---------------- kernel 2: QKV GEMM (8192x256 @ 256x768) -------
// Block tile 128(M) x 64(O); 8 waves each own a 32x32 tile (2x2 WMMA accs).
// Tiles staged with async global->LDS DMA; epilogue scatters to q (scaled),
// k, and transposed v.
__global__ void __launch_bounds__(256) vit_qkv(const u16* __restrict__ xn,
                                               const u16* __restrict__ wq,
                                               const float* __restrict__ qbias,
                                               u16* __restrict__ q,
                                               u16* __restrict__ k,
                                               u16* __restrict__ vt) {
    __shared__ u16 As[128 * 32];
    __shared__ u16 Bs[64 * 32];
    int t = threadIdx.x;
    int l = t & 31, w = t >> 5;
    int wm = w & 3, wn = w >> 2;
    int m0 = blockIdx.x * 128;
    int o0 = blockIdx.y * 64;
    int kb = (l >> 4) << 3;
    v8f acc[2][2] = {{{}, {}}, {{}, {}}};

    for (int kc = 0; kc < 8; ++kc) {
        __syncthreads();
        {   // A tile 128x32: 16 bf16 per thread, two async b128
            int r = t >> 1, c0 = (t & 1) * 16;
            const u16* g = xn + (size_t)(m0 + r) * CCH + kc * 32 + c0;
            async_cp16(As + r * 32 + c0,     g);
            async_cp16(As + r * 32 + c0 + 8, g + 8);
        }
        {   // B tile 64x32: 8 bf16 per thread, one async b128
            int r = t >> 2, c0 = (t & 3) * 8;
            async_cp16(Bs + r * 32 + c0,
                       wq + (size_t)(o0 + r) * CCH + kc * 32 + c0);
        }
        wait_async0();
        __syncthreads();
        v16bf af[2], bfr[2];
        #pragma unroll
        for (int s2 = 0; s2 < 2; ++s2)
            af[s2] = ldfrag(As + (wm * 32 + s2 * 16 + (l & 15)) * 32, kb);
        #pragma unroll
        for (int s2 = 0; s2 < 2; ++s2)
            bfr[s2] = ldfrag(Bs + (wn * 32 + s2 * 16 + (l & 15)) * 32, kb);
        #pragma unroll
        for (int sm = 0; sm < 2; ++sm)
            #pragma unroll
            for (int sn = 0; sn < 2; ++sn)
                acc[sm][sn] = __builtin_amdgcn_wmma_f32_16x16x32_bf16(
                    false, af[sm], false, bfr[sn], (short)0, acc[sm][sn], false, false);
    }
    // D layout: vgpr i -> row i + (l>>4)*8, col = l&15
    #pragma unroll
    for (int sm = 0; sm < 2; ++sm)
        #pragma unroll
        for (int sn = 0; sn < 2; ++sn) {
            int o = o0 + wn * 32 + sn * 16 + (l & 15);
            float bias = qbias[o];
            #pragma unroll
            for (int i = 0; i < 8; ++i) {
                int m = m0 + wm * 32 + sm * 16 + i + ((l >> 4) << 3);
                float val = acc[sm][sn][i] + bias;
                int bb = m >> 12, n = m & 4095;
                int c = o & 255, h = c >> 5, d = c & 31;
                size_t bh = (size_t)bb * NH + h;
                if (o < 256)      q[(bh * NTOK + n) * HD + d] = f2bf(val * QK_SCALE);
                else if (o < 512) k[(bh * NTOK + n) * HD + d] = f2bf(val);
                else              vt[(bh * HD + d) * NTOK + n] = f2bf(val);
            }
        }
}

// ---------------- kernel 3: flash attention ---------------------
// One block = 4 waves = 64 queries of one (b,h); loop over 64-key tiles.
// S = Q(16x32) @ K^T via WMMA, online softmax in f32 (D-row striping matches
// 16-lane shfl groups), P staged through LDS to re-layout D->A, O += P @ V.
__global__ void __launch_bounds__(128) vit_attn(const u16* __restrict__ q,
                                                const u16* __restrict__ k,
                                                const u16* __restrict__ vt,
                                                u16* __restrict__ ao) {
    __shared__ u16 Kt[64 * 32];       // [key][d]
    __shared__ u16 Vt[32 * 64];       // [d][key]
    __shared__ u16 Pb[4 * 16 * 64];   // per-wave P scratch
    int t = threadIdx.x, l = t & 31, w = t >> 5;
    int bh = blockIdx.y;
    int b = bh >> 3, h = bh & 7;
    int q0 = blockIdx.x * 64 + w * 16;
    int kb = (l >> 4) << 3;
    const u16* qp = q  + (size_t)bh * NTOK * HD;
    const u16* kp = k  + (size_t)bh * NTOK * HD;
    const u16* vp = vt + (size_t)bh * HD * NTOK;

    v16bf aQ = ldfrag(qp + (size_t)(q0 + (l & 15)) * HD, kb);
    v8f oacc[2] = {{}, {}};
    float m_i[8], l_i[8];
    #pragma unroll
    for (int i = 0; i < 8; ++i) { m_i[i] = -1e30f; l_i[i] = 0.f; }
    u16* Pw = Pb + w * 16 * 64;
    int rowoff = (l >> 4) << 3;

    for (int kt = 0; kt < NTOK / 64; ++kt) {
        __syncthreads();
        {   // K tile 64x32 (row-major by key): 2 async b128 per thread
            int r = t >> 1, c0 = (t & 1) * 16;
            const u16* g = kp + (size_t)(kt * 64 + r) * HD + c0;
            async_cp16(Kt + r * 32 + c0,     g);
            async_cp16(Kt + r * 32 + c0 + 8, g + 8);
        }
        {   // V^T tile 32x64 (row-major by d): 2 async b128 per thread
            int dd = t >> 2, c0 = (t & 3) * 16;
            const u16* g = vp + (size_t)dd * NTOK + kt * 64 + c0;
            async_cp16(Vt + dd * 64 + c0,     g);
            async_cp16(Vt + dd * 64 + c0 + 8, g + 8);
        }
        wait_async0();
        __syncthreads();

        v8f s4[4];
        #pragma unroll
        for (int ks = 0; ks < 4; ++ks) {
            v16bf bK = ldfrag(Kt + (ks * 16 + (l & 15)) * 32, kb);
            v8f zero = {};
            s4[ks] = __builtin_amdgcn_wmma_f32_16x16x32_bf16(
                false, aQ, false, bK, (short)0, zero, false, false);
        }
        // online softmax; row of vgpr i lives in a fixed 16-lane group
        #pragma unroll
        for (int i = 0; i < 8; ++i) {
            float mx = fmaxf(fmaxf(s4[0][i], s4[1][i]), fmaxf(s4[2][i], s4[3][i]));
            #pragma unroll
            for (int m = 1; m < 16; m <<= 1) mx = fmaxf(mx, __shfl_xor(mx, m));
            float mn = fmaxf(m_i[i], mx);
            float alpha = exp2f((m_i[i] - mn) * LOG2E);
            float rs = 0.f;
            #pragma unroll
            for (int ks = 0; ks < 4; ++ks) {
                float p = exp2f((s4[ks][i] - mn) * LOG2E);
                s4[ks][i] = p;
                rs += p;
            }
            #pragma unroll
            for (int m = 1; m < 16; m <<= 1) rs += __shfl_xor(rs, m);
            l_i[i] = l_i[i] * alpha + rs;
            m_i[i] = mn;
            oacc[0][i] = oacc[0][i] * alpha;
            oacc[1][i] = oacc[1][i] * alpha;
        }
        // stage P (D layout) to LDS, re-read as A fragments
        #pragma unroll
        for (int ks = 0; ks < 4; ++ks)
            #pragma unroll
            for (int i = 0; i < 8; ++i)
                Pw[(i + rowoff) * 64 + ks * 16 + (l & 15)] = f2bf(s4[ks][i]);
        #pragma unroll
        for (int kc = 0; kc < 2; ++kc) {
            v16bf aP = ldfrag(Pw + (l & 15) * 64 + kc * 32, kb);
            #pragma unroll
            for (int dn = 0; dn < 2; ++dn) {
                v16bf bV = ldfrag(Vt + (dn * 16 + (l & 15)) * 64 + kc * 32, kb);
                oacc[dn] = __builtin_amdgcn_wmma_f32_16x16x32_bf16(
                    false, aP, false, bV, (short)0, oacc[dn], false, false);
            }
        }
    }
    // normalize and write (B, N, C) bf16 for the proj GEMM
    #pragma unroll
    for (int dn = 0; dn < 2; ++dn)
        #pragma unroll
        for (int i = 0; i < 8; ++i) {
            int qi = q0 + i + rowoff;
            float val = oacc[dn][i] / l_i[i];
            ao[((size_t)b * NTOK + qi) * CCH + h * HD + dn * 16 + (l & 15)] = f2bf(val);
        }
}

// ---------------- kernel 4: proj GEMM + gamma*attn + residual ---
__global__ void __launch_bounds__(256) vit_proj(const u16* __restrict__ ao,
                                                const u16* __restrict__ wp,
                                                const float* __restrict__ pbias,
                                                const float* __restrict__ gamma,
                                                const float* __restrict__ x,
                                                float* __restrict__ out) {
    __shared__ u16 As[128 * 32];
    __shared__ u16 Bs[64 * 32];
    int t = threadIdx.x;
    int l = t & 31, w = t >> 5;
    int wm = w & 3, wn = w >> 2;
    int m0 = blockIdx.x * 128;
    int o0 = blockIdx.y * 64;
    int kb = (l >> 4) << 3;
    v8f acc[2][2] = {{{}, {}}, {{}, {}}};

    for (int kc = 0; kc < 8; ++kc) {
        __syncthreads();
        {
            int r = t >> 1, c0 = (t & 1) * 16;
            const u16* g = ao + (size_t)(m0 + r) * CCH + kc * 32 + c0;
            async_cp16(As + r * 32 + c0,     g);
            async_cp16(As + r * 32 + c0 + 8, g + 8);
        }
        {
            int r = t >> 2, c0 = (t & 3) * 8;
            async_cp16(Bs + r * 32 + c0,
                       wp + (size_t)(o0 + r) * CCH + kc * 32 + c0);
        }
        wait_async0();
        __syncthreads();
        v16bf af[2], bfr[2];
        #pragma unroll
        for (int s2 = 0; s2 < 2; ++s2)
            af[s2] = ldfrag(As + (wm * 32 + s2 * 16 + (l & 15)) * 32, kb);
        #pragma unroll
        for (int s2 = 0; s2 < 2; ++s2)
            bfr[s2] = ldfrag(Bs + (wn * 32 + s2 * 16 + (l & 15)) * 32, kb);
        #pragma unroll
        for (int sm = 0; sm < 2; ++sm)
            #pragma unroll
            for (int sn = 0; sn < 2; ++sn)
                acc[sm][sn] = __builtin_amdgcn_wmma_f32_16x16x32_bf16(
                    false, af[sm], false, bfr[sn], (short)0, acc[sm][sn], false, false);
    }
    float g = gamma[0];
    #pragma unroll
    for (int sm = 0; sm < 2; ++sm)
        #pragma unroll
        for (int sn = 0; sn < 2; ++sn) {
            int o = o0 + wn * 32 + sn * 16 + (l & 15);
            float bias = pbias[o];
            #pragma unroll
            for (int i = 0; i < 8; ++i) {
                int m = m0 + wm * 32 + sm * 16 + i + ((l >> 4) << 3);
                int bb = m >> 12, n = m & 4095;
                size_t idx = ((size_t)bb * CCH + o) * NTOK + n;
                out[idx] = g * (acc[sm][sn][i] + bias) + x[idx];
            }
        }
}

extern "C" void kernel_launch(void* const* d_in, const int* in_sizes, int n_in,
                              void* d_out, int out_size, void* d_ws, size_t ws_size,
                              hipStream_t stream) {
    const float* x      = (const float*)d_in[0];
    const float* norm_w = (const float*)d_in[1];
    const float* norm_b = (const float*)d_in[2];
    const float* qkv_w  = (const float*)d_in[3];
    const float* qkv_b  = (const float*)d_in[4];
    const float* proj_w = (const float*)d_in[5];
    const float* proj_b = (const float*)d_in[6];
    const float* gamma  = (const float*)d_in[7];
    float* out = (float*)d_out;

    char* ws = (char*)d_ws;
    const size_t MB = 1024 * 1024;
    u16* xn   = (u16*)(ws);             // 8192*256 bf16 = 4 MB
    u16* qb16 = (u16*)(ws + 4 * MB);
    u16* kb16 = (u16*)(ws + 8 * MB);
    u16* vt16 = (u16*)(ws + 12 * MB);
    u16* ao   = (u16*)(ws + 16 * MB);
    u16* wqb  = (u16*)(ws + 20 * MB);                  // 768*256 bf16
    u16* wpb  = (u16*)(ws + 20 * MB + 768 * 256 * 2);  // 256*256 bf16

    vit_cvt_w<<<768, 256, 0, stream>>>(qkv_w, proj_w, wqb, wpb);
    vit_ln<<<256, 256, 0, stream>>>(x, norm_w, norm_b, xn);
    vit_qkv<<<dim3(64, 12), 256, 0, stream>>>(xn, wqb, qkv_b, qb16, kb16, vt16);
    vit_attn<<<dim3(64, 16), 128, 0, stream>>>(qb16, kb16, vt16, ao);
    vit_proj<<<dim3(64, 4), 256, 0, stream>>>(ao, wpb, proj_b, gamma, x, out);
}
